// Pool_8143257993643
// MI455X (gfx1250) — matrix operand order repaired and checked by reading, not compiled, not run on hardware
//
#include <hip/hip_runtime.h>
#include <hip/hip_bf16.h>
#include <stdint.h>

#define NN 4096
#define DD 256
#define KK 2048
#define EPS 1e-5f

typedef __attribute__((ext_vector_type(16))) __bf16 v16bf;
typedef __attribute__((ext_vector_type(8)))  float  v8f;

union V16BF { unsigned short u[16]; v16bf v; };

__device__ __forceinline__ unsigned short f2bf(float f) {
  unsigned u = __float_as_uint(f);
  return (unsigned short)((u + 0x7FFFu + ((u >> 16) & 1u)) >> 16);
}
__device__ __forceinline__ float bf2f(unsigned short b) {
  return __uint_as_float(((unsigned)b) << 16);
}
__device__ __forceinline__ float sigmoidf_(float x) {
  return 1.0f / (1.0f + __expf(-x));
}

// ---- CDNA5 async global->LDS copy (ASYNCcnt path), inline asm ----
__device__ __forceinline__ void async_copy_b128(unsigned lds_off, const void* gaddr) {
  asm volatile("global_load_async_to_lds_b128 %0, %1, off"
               :: "v"(lds_off), "v"(gaddr) : "memory");
}
__device__ __forceinline__ void wait_async0() {
  asm volatile("s_wait_asynccnt 0" ::: "memory");
}

// stage a K=64 x N=16 bf16 tile (2KB) into LDS, wave0 only (4 async b128/lane)
__device__ __forceinline__ void stage_bf16_tile(unsigned bufbase, const unsigned short* src,
                                                int k, int nbase, int ldb, int lane) {
  int r = lane >> 1, hh = lane & 1;
  #pragma unroll
  for (int j = 0; j < 4; j++) {
    async_copy_b128(bufbase + (unsigned)((j * 16 + r) * 32 + hh * 16),
                    src + (size_t)(k + j * 16 + r) * ldb + nbase + hh * 8);
  }
}
// stage a K=64 x N=16 f32 tile (4KB) into LDS, wave0 only (8 async b128/lane)
__device__ __forceinline__ void stage_f32_tile(unsigned bufbase, const float* src,
                                               int k, int nbase, int ldb, int lane) {
  int r = lane >> 2, q = lane & 3;
  #pragma unroll
  for (int j = 0; j < 8; j++) {
    async_copy_b128(bufbase + (unsigned)((j * 8 + r) * 64 + q * 16),
                    src + (size_t)(k + j * 8 + r) * ldb + nbase + q * 4);
  }
}

// ---------------- column stats: mu, 1/sqrt(var+eps) ----------------
__global__ void colstats_kernel(const float* __restrict__ h, float* mu, float* rsig) {
  __shared__ float s1[256], s2[256];
  int d = blockIdx.x, t = threadIdx.x;
  float s = 0.f, q = 0.f;
  for (int n = t; n < NN; n += 256) { float v = h[(size_t)n * DD + d]; s += v; q += v * v; }
  s1[t] = s; s2[t] = q; __syncthreads();
  for (int o = 128; o > 0; o >>= 1) {
    if (t < o) { s1[t] += s1[t + o]; s2[t] += s2[t + o]; }
    __syncthreads();
  }
  if (t == 0) {
    float m = s1[0] / (float)NN;
    float var = s2[0] / (float)NN - m * m;
    mu[d] = m; rsig[d] = rsqrtf(var + EPS);
  }
}

// ---------------- normalize h -> h_norm (f32) + h_bf16 ----------------
__global__ void norm_kernel(const float* __restrict__ h, const float* mu, const float* rsig,
                            const float* gamma, const float* beta,
                            float* h_norm, unsigned short* h_bf) {
  int n = blockIdx.x, d = threadIdx.x;
  float v = (h[(size_t)n * DD + d] - mu[d]) * rsig[d] * gamma[d] + beta[d];
  h_norm[(size_t)n * DD + d] = v;
  h_bf[(size_t)n * DD + d] = f2bf(v);
}

// ---------------- degree (and 1/degree) = row sums of g ----------------
__global__ void deg_kernel(const float* __restrict__ g, float* deg, float* invdeg) {
  __shared__ float s[256];
  int n = blockIdx.x, t = threadIdx.x;
  float a = 0.f;
  for (int j = t; j < NN; j += 256) a += (g[(size_t)n * NN + j] != 0.f) ? 1.f : 0.f;
  s[t] = a; __syncthreads();
  for (int o = 128; o > 0; o >>= 1) { if (t < o) s[t] += s[t + o]; __syncthreads(); }
  if (t == 0) { deg[n] = s[0]; invdeg[n] = 1.f / s[0]; }
}

// ---------------- GEMM: agg = (g @ h_norm) * invdeg  [bf16 WMMA] ----------------
// 8 waves/block: 128(M) x 16(N) tile; shared double-buffered B slice (K=64),
// staged by wave0 via async global->LDS; 2 independent accumulators break
// the WMMA->WMMA RAW chain.
__global__ void __launch_bounds__(256) gemm_agg_kernel(
    const float* __restrict__ g, const unsigned short* __restrict__ hb,
    const float* __restrict__ invdeg, float* __restrict__ agg) {
  __shared__ __align__(16) unsigned short bs[2][64 * 16]; // 2 x 2KB
  int lane = threadIdx.x & 31, wid = threadIdx.x >> 5;
  int mtile = blockIdx.x * 128 + wid * 16;
  int nbase = blockIdx.y * 16;
  int m = lane & 15, hi = lane >> 4, klo = hi * 8;
  v8f acc0 = {}, acc1 = {};
  unsigned base[2] = { (unsigned)(uintptr_t)(void*)&bs[0][0],
                       (unsigned)(uintptr_t)(void*)&bs[1][0] };
  if (wid == 0) stage_bf16_tile(base[0], hb, 0, nbase, DD, lane);
  const float* arowbase = g + (size_t)(mtile + m) * NN;
  for (int it = 0; it < NN / 64; ++it) {
    int k = it * 64, buf = it & 1;
    if (wid == 0) wait_async0();
    __syncthreads();
    if (wid == 0 && it + 1 < NN / 64)
      stage_bf16_tile(base[buf ^ 1], hb, k + 64, nbase, DD, lane);
    if (k + 64 < NN) __builtin_prefetch(arowbase + k + 64 + klo, 0, 1);
    #pragma unroll
    for (int s = 0; s < 2; ++s) {
      const float* arow = arowbase + k + s * 32 + klo;
      float4 a0 = *(const float4*)(arow);
      float4 a1 = *(const float4*)(arow + 4);
      float4 a2 = *(const float4*)(arow + 16);
      float4 a3 = *(const float4*)(arow + 20);
      V16BF A;
      A.u[0]=f2bf(a0.x); A.u[1]=f2bf(a0.y); A.u[2]=f2bf(a0.z); A.u[3]=f2bf(a0.w);
      A.u[4]=f2bf(a1.x); A.u[5]=f2bf(a1.y); A.u[6]=f2bf(a1.z); A.u[7]=f2bf(a1.w);
      A.u[8]=f2bf(a2.x); A.u[9]=f2bf(a2.y); A.u[10]=f2bf(a2.z); A.u[11]=f2bf(a2.w);
      A.u[12]=f2bf(a3.x); A.u[13]=f2bf(a3.y); A.u[14]=f2bf(a3.z); A.u[15]=f2bf(a3.w);
      V16BF B;
      int kb = s * 32 + hi * 8;
      #pragma unroll
      for (int v = 0; v < 8; v++) {
        B.u[2 * v]     = bs[buf][(kb + 2 * v + (v >= 4 ? 8 : 0)) * 16 + m];
        B.u[2 * v + 1] = bs[buf][(kb + 2 * v + 1 + (v >= 4 ? 8 : 0)) * 16 + m];
      }
      if (s == 0)
        acc0 = __builtin_amdgcn_wmma_f32_16x16x32_bf16(false, A.v, false, B.v,
                                                       (short)0, acc0, false, false);
      else
        acc1 = __builtin_amdgcn_wmma_f32_16x16x32_bf16(false, A.v, false, B.v,
                                                       (short)0, acc1, false, false);
    }
    __syncthreads();
  }
  #pragma unroll
  for (int v = 0; v < 8; v++) {
    int row = mtile + v + hi * 8;
    int col = nbase + m;
    agg[(size_t)row * DD + col] = (acc0[v] + acc1[v]) * invdeg[row];
  }
}

// ---------------- per-row features: Z1->pl, Z3, pf ----------------
__global__ void rowfeat_kernel(const float* __restrict__ h_norm, const float* __restrict__ agg,
                               const float* __restrict__ deg, const float* __restrict__ w,
                               const float* __restrict__ b,
                               float* Z3, float* pl, float* pf) {
  __shared__ float sa[256], sb[256], sc[256];
  int n = blockIdx.x, t = threadIdx.x;
  float hv = h_norm[(size_t)n * DD + t];
  float av = agg[(size_t)n * DD + t];
  float wv = w[t];
  sa[t] = fabsf(hv - av); sb[t] = av * wv; sc[t] = hv * wv;
  __syncthreads();
  for (int o = 128; o > 0; o >>= 1) {
    if (t < o) { sa[t] += sa[t + o]; sb[t] += sb[t + o]; sc[t] += sc[t + o]; }
    __syncthreads();
  }
  if (t == 0) {
    pl[n] = sigmoidf_(sa[0] + deg[n]);
    Z3[n] = sb[0] + b[0];
    pf[n] = sigmoidf_(sc[0] + b[0]);
  }
}

// ---------------- softmax reduction over Z3 ----------------
__global__ void softmax_reduce_kernel(const float* __restrict__ Z3, float* red) {
  __shared__ float s[256];
  int t = threadIdx.x;
  float mx = -3.4e38f;
  for (int i = t; i < NN; i += 256) mx = fmaxf(mx, Z3[i]);
  s[t] = mx; __syncthreads();
  for (int o = 128; o > 0; o >>= 1) { if (t < o) s[t] = fmaxf(s[t], s[t + o]); __syncthreads(); }
  mx = s[0]; __syncthreads();
  float sum = 0.f;
  for (int i = t; i < NN; i += 256) sum += __expf(Z3[i] - mx);
  s[t] = sum; __syncthreads();
  for (int o = 128; o > 0; o >>= 1) { if (t < o) s[t] += s[t + o]; __syncthreads(); }
  if (t == 0) { red[0] = mx; red[1] = s[0]; }
}

// ---------------- combine scores ----------------
__global__ void scores_kernel(const float* Z3, const float* pl, const float* pf,
                              const float* red, const float* sigma1, float* scores) {
  int i = blockIdx.x * 256 + threadIdx.x;
  float pg = __expf(Z3[i] - red[0]) / red[1];
  float pt = sigmoidf_(pl[i] + pg);
  float s1 = sigma1[0];
  scores[i] = s1 * pt + (1.f - s1) * pf[i];
}

// ---------------- exact top-k via rank counting (stable, low-index-first ties) ----
__global__ void topk_kernel(const float* __restrict__ scores, int* idx) {
  __shared__ int s[256];
  int i = blockIdx.x, t = threadIdx.x;
  float si = scores[i];
  int c = 0;
  for (int j = t; j < NN; j += 256) {
    float sj = scores[j];
    c += (sj > si) || (sj == si && j < i);
  }
  s[t] = c; __syncthreads();
  for (int o = 128; o > 0; o >>= 1) { if (t < o) s[t] += s[t + o]; __syncthreads(); }
  if (t == 0) { int r = s[0]; if (r < KK) idx[r] = i; }
}

// ---------------- h_att = scores[:,None] * h_norm  (bf16) ----------------
__global__ void hatt_kernel(const float* __restrict__ h_norm, const float* __restrict__ scores,
                            unsigned short* hatt) {
  int n = blockIdx.x, d = threadIdx.x;
  hatt[(size_t)n * DD + d] = f2bf(scores[n] * h_norm[(size_t)n * DD + d]);
}

// ---------------- a_hat rows = ((g[idx] @ g) != 0)  [bf16 WMMA, f32 counts] ----
__global__ void __launch_bounds__(256) gemm_twohop_kernel(
    const float* __restrict__ g, const int* __restrict__ idx,
    unsigned short* __restrict__ ahat) {
  __shared__ __align__(16) float bsf[2][64 * 16]; // 2 x 4KB
  int lane = threadIdx.x & 31, wid = threadIdx.x >> 5;
  int mtile = blockIdx.x * 128 + wid * 16;
  int nbase = blockIdx.y * 16;
  int m = lane & 15, hi = lane >> 4, klo = hi * 8;
  int grow = idx[mtile + m];
  v8f acc0 = {}, acc1 = {};
  unsigned base[2] = { (unsigned)(uintptr_t)(void*)&bsf[0][0],
                       (unsigned)(uintptr_t)(void*)&bsf[1][0] };
  if (wid == 0) stage_f32_tile(base[0], g, 0, nbase, NN, lane);
  const float* arowbase = g + (size_t)grow * NN;
  for (int it = 0; it < NN / 64; ++it) {
    int k = it * 64, buf = it & 1;
    if (wid == 0) wait_async0();
    __syncthreads();
    if (wid == 0 && it + 1 < NN / 64)
      stage_f32_tile(base[buf ^ 1], g, k + 64, nbase, NN, lane);
    if (k + 64 < NN) __builtin_prefetch(arowbase + k + 64 + klo, 0, 1);
    #pragma unroll
    for (int s = 0; s < 2; ++s) {
      const float* arow = arowbase + k + s * 32 + klo;
      float4 a0 = *(const float4*)(arow);
      float4 a1 = *(const float4*)(arow + 4);
      float4 a2 = *(const float4*)(arow + 16);
      float4 a3 = *(const float4*)(arow + 20);
      V16BF A;
      A.u[0]=(a0.x!=0.f)?0x3F80:0; A.u[1]=(a0.y!=0.f)?0x3F80:0;
      A.u[2]=(a0.z!=0.f)?0x3F80:0; A.u[3]=(a0.w!=0.f)?0x3F80:0;
      A.u[4]=(a1.x!=0.f)?0x3F80:0; A.u[5]=(a1.y!=0.f)?0x3F80:0;
      A.u[6]=(a1.z!=0.f)?0x3F80:0; A.u[7]=(a1.w!=0.f)?0x3F80:0;
      A.u[8]=(a2.x!=0.f)?0x3F80:0; A.u[9]=(a2.y!=0.f)?0x3F80:0;
      A.u[10]=(a2.z!=0.f)?0x3F80:0; A.u[11]=(a2.w!=0.f)?0x3F80:0;
      A.u[12]=(a3.x!=0.f)?0x3F80:0; A.u[13]=(a3.y!=0.f)?0x3F80:0;
      A.u[14]=(a3.z!=0.f)?0x3F80:0; A.u[15]=(a3.w!=0.f)?0x3F80:0;
      V16BF B;
      int kb = s * 32 + hi * 8;
      #pragma unroll
      for (int v = 0; v < 8; v++) {
        float b0 = bsf[buf][(kb + 2 * v + (v >= 4 ? 8 : 0)) * 16 + m];
        float b1 = bsf[buf][(kb + 2 * v + 1 + (v >= 4 ? 8 : 0)) * 16 + m];
        B.u[2 * v]     = (b0 != 0.f) ? 0x3F80 : 0;
        B.u[2 * v + 1] = (b1 != 0.f) ? 0x3F80 : 0;
      }
      if (s == 0)
        acc0 = __builtin_amdgcn_wmma_f32_16x16x32_bf16(false, A.v, false, B.v,
                                                       (short)0, acc0, false, false);
      else
        acc1 = __builtin_amdgcn_wmma_f32_16x16x32_bf16(false, A.v, false, B.v,
                                                       (short)0, acc1, false, false);
    }
    __syncthreads();
  }
  #pragma unroll
  for (int v = 0; v < 8; v++) {
    int row = mtile + v + hi * 8;
    int col = nbase + m;
    ahat[(size_t)row * NN + col] = ((acc0[v] + acc1[v]) != 0.f) ? 0x3F80 : 0;
  }
}

// ---------------- new_h = a_hat @ h_att  [bf16 WMMA] ----------------
__global__ void __launch_bounds__(256) gemm_newh_kernel(
    const unsigned short* __restrict__ ahat, const unsigned short* __restrict__ hatt,
    float* __restrict__ outh) {
  __shared__ __align__(16) unsigned short bs[2][64 * 16];
  int lane = threadIdx.x & 31, wid = threadIdx.x >> 5;
  int mtile = blockIdx.x * 128 + wid * 16;
  int nbase = blockIdx.y * 16;
  int m = lane & 15, hi = lane >> 4, klo = hi * 8;
  v8f acc0 = {}, acc1 = {};
  unsigned base[2] = { (unsigned)(uintptr_t)(void*)&bs[0][0],
                       (unsigned)(uintptr_t)(void*)&bs[1][0] };
  if (wid == 0) stage_bf16_tile(base[0], hatt, 0, nbase, DD, lane);
  const unsigned short* arowbase = ahat + (size_t)(mtile + m) * NN;
  for (int it = 0; it < NN / 64; ++it) {
    int k = it * 64, buf = it & 1;
    if (wid == 0) wait_async0();
    __syncthreads();
    if (wid == 0 && it + 1 < NN / 64)
      stage_bf16_tile(base[buf ^ 1], hatt, k + 64, nbase, DD, lane);
    if (k + 64 < NN) __builtin_prefetch(arowbase + k + 64 + klo, 0, 1);
    #pragma unroll
    for (int s = 0; s < 2; ++s) {
      const unsigned short* arow = arowbase + k + s * 32 + klo;
      uint4 p0 = *(const uint4*)(arow);        // K = klo..klo+7
      uint4 p1 = *(const uint4*)(arow + 16);   // K = klo+16..klo+23
      V16BF A;
      A.u[0]=(unsigned short)(p0.x); A.u[1]=(unsigned short)(p0.x>>16);
      A.u[2]=(unsigned short)(p0.y); A.u[3]=(unsigned short)(p0.y>>16);
      A.u[4]=(unsigned short)(p0.z); A.u[5]=(unsigned short)(p0.z>>16);
      A.u[6]=(unsigned short)(p0.w); A.u[7]=(unsigned short)(p0.w>>16);
      A.u[8]=(unsigned short)(p1.x); A.u[9]=(unsigned short)(p1.x>>16);
      A.u[10]=(unsigned short)(p1.y); A.u[11]=(unsigned short)(p1.y>>16);
      A.u[12]=(unsigned short)(p1.z); A.u[13]=(unsigned short)(p1.z>>16);
      A.u[14]=(unsigned short)(p1.w); A.u[15]=(unsigned short)(p1.w>>16);
      V16BF B;
      int kb = s * 32 + hi * 8;
      #pragma unroll
      for (int v = 0; v < 8; v++) {
        B.u[2 * v]     = bs[buf][(kb + 2 * v + (v >= 4 ? 8 : 0)) * 16 + m];
        B.u[2 * v + 1] = bs[buf][(kb + 2 * v + 1 + (v >= 4 ? 8 : 0)) * 16 + m];
      }
      if (s == 0)
        acc0 = __builtin_amdgcn_wmma_f32_16x16x32_bf16(false, A.v, false, B.v,
                                                       (short)0, acc0, false, false);
      else
        acc1 = __builtin_amdgcn_wmma_f32_16x16x32_bf16(false, A.v, false, B.v,
                                                       (short)0, acc1, false, false);
    }
    __syncthreads();
  }
  #pragma unroll
  for (int v = 0; v < 8; v++) {
    int row = mtile + v + hi * 8;
    int col = nbase + m;
    outh[(size_t)row * DD + col] = acc0[v] + acc1[v];
  }
}

// ---------------- g_new = row-normalized a_hat[:, idx]; also emit idx ----------------
__global__ void gnew_kernel(const unsigned short* __restrict__ ahat,
                            const int* __restrict__ idx,
                            float* __restrict__ gnew, int* __restrict__ out_idx) {
  __shared__ float s[256];
  __shared__ float inv;
  int r = blockIdx.x, t = threadIdx.x;
  float vals[8];
  float a = 0.f;
  #pragma unroll
  for (int j = 0; j < 8; j++) {
    int c = t + 256 * j;
    float v = bf2f(ahat[(size_t)r * NN + idx[c]]);
    vals[j] = v; a += v;
  }
  s[t] = a; __syncthreads();
  for (int o = 128; o > 0; o >>= 1) { if (t < o) s[t] += s[t + o]; __syncthreads(); }
  if (t == 0) { inv = 1.f / s[0]; out_idx[r] = idx[r]; }
  __syncthreads();
  #pragma unroll
  for (int j = 0; j < 8; j++) {
    int c = t + 256 * j;
    gnew[(size_t)r * KK + c] = vals[j] * inv;
  }
}

extern "C" void kernel_launch(void* const* d_in, const int* in_sizes, int n_in,
                              void* d_out, int out_size, void* d_ws, size_t ws_size,
                              hipStream_t stream) {
  const float* g      = (const float*)d_in[0];
  const float* h      = (const float*)d_in[1];
  const float* gamma  = (const float*)d_in[2];
  const float* beta   = (const float*)d_in[3];
  const float* w_proj = (const float*)d_in[4];
  const float* b_proj = (const float*)d_in[5];
  const float* sigma1 = (const float*)d_in[6];

  // workspace layout
  float* mu     = (float*)d_ws;            // 256
  float* rsig   = mu + 256;                // 256
  float* deg    = rsig + 256;              // 4096
  float* invdeg = deg + NN;                // 4096
  float* Z3     = invdeg + NN;             // 4096
  float* pl     = Z3 + NN;                 // 4096
  float* pf     = pl + NN;                 // 4096
  float* scores = pf + NN;                 // 4096
  float* red    = scores + NN;             // 8
  int*   idx    = (int*)(red + 8);         // 2048
  char*  big    = (char*)d_ws + (1u << 17);           // 128KB
  float* h_norm = (float*)big;                         // 4MB
  float* agg    = h_norm + (size_t)NN * DD;            // 4MB
  unsigned short* h_bf = (unsigned short*)(agg + (size_t)NN * DD);  // 2MB
  unsigned short* hatt = h_bf + (size_t)NN * DD;                     // 2MB
  unsigned short* ahat = hatt + (size_t)NN * DD;                     // 16MB

  float* out_gnew = (float*)d_out;                              // 2048*2048
  float* out_newh = out_gnew + (size_t)KK * KK;                 // 2048*256
  int*   out_idx  = (int*)(out_newh + (size_t)KK * DD);         // 2048

  colstats_kernel<<<DD, 256, 0, stream>>>(h, mu, rsig);
  norm_kernel<<<NN, 256, 0, stream>>>(h, mu, rsig, gamma, beta, h_norm, h_bf);
  deg_kernel<<<NN, 256, 0, stream>>>(g, deg, invdeg);
  gemm_agg_kernel<<<dim3(NN / 128, DD / 16), 256, 0, stream>>>(g, h_bf, invdeg, agg);
  rowfeat_kernel<<<NN, 256, 0, stream>>>(h_norm, agg, deg, w_proj, b_proj, Z3, pl, pf);
  softmax_reduce_kernel<<<1, 256, 0, stream>>>(Z3, red);
  scores_kernel<<<NN / 256, 256, 0, stream>>>(Z3, pl, pf, red, sigma1, scores);
  topk_kernel<<<NN, 256, 0, stream>>>(scores, idx);
  hatt_kernel<<<NN, 256, 0, stream>>>(h_norm, scores, hatt);
  gemm_twohop_kernel<<<dim3(KK / 128, NN / 16), 256, 0, stream>>>(g, idx, ahat);
  gemm_newh_kernel<<<dim3(KK / 128, DD / 16), 256, 0, stream>>>(ahat, hatt, out_newh);
  gnew_kernel<<<KK, 256, 0, stream>>>(ahat, idx, out_gnew, out_idx);
}
